// TDTNet_5025111736708
// MI455X (gfx1250) — matrix operand order repaired and checked by reading, not compiled, not run on hardware
//
#include <hip/hip_runtime.h>
#include <hip/hip_bf16.h>

typedef __attribute__((ext_vector_type(16))) _Float16 v16h;
typedef __attribute__((ext_vector_type(8)))  float    v8f;

#define NB    16
#define NN    512
#define KNB   48
#define ROWS  64      // K+1 = 49 padded to 64 (4 M-tiles of 16)
#define DD    128
#define GG    32
#define NH    8
#define HD    16
#define CUTR  5.0f

// ---------------------------------------------------------------------------
// WMMA fragment helpers — A/B in contiguous k order (this code's choice), C/D as hardware (wave32)
// ---------------------------------------------------------------------------

// A fragment: 16x32 f16, row-major source with leading dim `ld`.
// VGPR v (0..7): K = k0 + half*16 + 2v,2v+1 (contiguous, matches B).
__device__ __forceinline__ v16h load_a_frag(const _Float16* __restrict__ src,
                                            int ld, int row0, int k0, int lane) {
    const int m    = lane & 15;
    const int half = lane >> 4;
    const _Float16* row = src + (row0 + m) * ld + k0 + half * 16;
    v16h a;
#pragma unroll
    for (int v = 0; v < 4; ++v) {
        a[2 * v]     = row[2 * v];
        a[2 * v + 1] = row[2 * v + 1];
    }
#pragma unroll
    for (int v = 0; v < 4; ++v) {
        a[8 + 2 * v] = row[8 + 2 * v];
        a[9 + 2 * v] = row[8 + 2 * v + 1];
    }
    return a;
}

// B fragment: 32x16 f16. Source is W^T (row = output col n, contiguous in k).
// lanes 0-15 hold K = k0..k0+15 of column n; lanes 16-31 hold K = k0+16..k0+31.
__device__ __forceinline__ v16h load_b_frag(const _Float16* __restrict__ WT,
                                            int ld, int n0, int k0, int lane) {
    const int n    = n0 + (lane & 15);
    const int half = lane >> 4;
    return *(const v16h*)(WT + n * ld + k0 + half * 16);
}

// C/D fragment: 16x16 f32. VGPR v holds element [row0 + v + 8*half][n0 + lane%16].
__device__ __forceinline__ void store_c_frag(float* __restrict__ dst, int ld,
                                             int row0, int n0, int lane, v8f c) {
    const int n    = n0 + (lane & 15);
    const int half = lane >> 4;
#pragma unroll
    for (int v = 0; v < 8; ++v)
        dst[(row0 + v + 8 * half) * ld + n] = c[v];
}

// Async gather of 16 bytes from global into LDS (tracked by ASYNCcnt).
__device__ __forceinline__ void async_copy_b128(unsigned lds_off,
                                                const float* gaddr) {
    asm volatile("global_load_async_to_lds_b128 %0, %1, off"
                 :: "v"(lds_off), "v"(gaddr) : "memory");
}
__device__ __forceinline__ void wait_async0() {
    asm volatile("s_wait_asynccnt 0x0" ::: "memory");
}

// ---------------------------------------------------------------------------
// Prep: convert Wk, Wv (128x128) and filt_W (32x128) to transposed f16 in ws
// ---------------------------------------------------------------------------
__global__ void prep_weights(const float* __restrict__ Wk,
                             const float* __restrict__ Wv,
                             const float* __restrict__ fW,
                             _Float16* __restrict__ WkT,
                             _Float16* __restrict__ WvT,
                             _Float16* __restrict__ fWt) {
    int t = blockIdx.x * blockDim.x + threadIdx.x;
    if (t < DD * DD) {
        int n = t >> 7, k = t & 127;                 // WT[n][k] = W[k][n]
        WkT[n * DD + k] = (_Float16)Wk[k * DD + n];
        WvT[n * DD + k] = (_Float16)Wv[k * DD + n];
    }
    if (t < DD * GG) {
        int n = t / GG, g = t % GG;
        fWt[n * GG + g] = (_Float16)fW[g * DD + n];
    }
}

// ---------------------------------------------------------------------------
// Fused per-atom kernel: 1 block = 1 atom, 256 threads = 8 wave32
// ---------------------------------------------------------------------------
__launch_bounds__(256)
__global__ void tdtnet_fused(const float* __restrict__ positions,
                             const int*   __restrict__ z,
                             const int*   __restrict__ neighbors,
                             const float* __restrict__ nmask,
                             const float* __restrict__ embedding,
                             const float* __restrict__ filt_b,
                             const float* __restrict__ Wq,
                             const float* __restrict__ Wo,
                             const _Float16* __restrict__ WkT,
                             const _Float16* __restrict__ WvT,
                             const _Float16* __restrict__ fWt,
                             float* __restrict__ out) {
    __shared__ float    s_x[DD];
    __shared__ float    s_q[DD];
    __shared__ float    s_riij[ROWS];
    __shared__ float    s_cut[ROWS];
    __shared__ float    s_mask[ROWS];
    __shared__ int      s_zrow[ROWS];
    __shared__ float    s_nodes[ROWS * DD];   // gathered node embeddings (async)
    __shared__ _Float16 s_fA[ROWS * GG];      // gaussian features (A of filter GEMM)
    __shared__ _Float16 s_kv[ROWS * DD];      // kv_in, f16 (A of K/V projections)
    __shared__ float    s_kp[ROWS * DD];      // K projection
    __shared__ float    s_vp[ROWS * DD];      // V projection
    __shared__ float    s_sc[NH * ROWS];      // scores / attn
    __shared__ float    s_inv[NH];
    __shared__ float    s_agg[DD];

    const int tid  = threadIdx.x;
    const int lane = tid & 31;
    const int wave = tid >> 5;
    const int atom = blockIdx.x;             // b*NN + n
    const int b    = atom / NN;

    // ---- stage 0: self embedding, neighbor distances, cutoff, mask --------
    const int zi = z[atom];
    const float px = positions[atom * 3 + 0];
    const float py = positions[atom * 3 + 1];
    const float pz = positions[atom * 3 + 2];

    if (tid < DD) s_x[tid] = embedding[zi * DD + tid];
    if (tid < ROWS) {
        float r;
        if (tid == 0) {
            r = 0.01f; s_zrow[0] = zi; s_mask[0] = 1.0f;
        } else if (tid <= KNB) {
            const int nb   = neighbors[atom * KNB + tid - 1];
            const int nidx = b * NN + nb;
            const float dx = positions[nidx * 3 + 0] - px;
            const float dy = positions[nidx * 3 + 1] - py;
            const float dz = positions[nidx * 3 + 2] - pz;
            r = sqrtf(dx * dx + dy * dy + dz * dz + 1e-12f);
            s_zrow[tid] = z[nidx];
            s_mask[tid] = nmask[atom * KNB + tid - 1];
        } else {                             // padded rows 49..63
            r = 1.0f; s_zrow[tid] = 0; s_mask[tid] = 0.0f;
        }
        s_riij[tid] = r;
        float c = 0.0f;
        if (tid <= KNB && r < CUTR)
            c = 0.5f * (__cosf(r * 0.62831853f) + 1.0f);   // pi/5
        s_cut[tid] = c;                       // zero for padded rows
    }
    __syncthreads();

    // ---- stage 1a: async prefetch of node embeddings into LDS --------------
    // 49 rows x 512B, 16B chunks: 1568 transfers, overlapped with VALU below.
    {
        const unsigned lds_base =
            (unsigned)(size_t)(__attribute__((address_space(3))) float*)s_nodes;
        for (int e = tid; e < (KNB + 1) * 32; e += 256) {
            const int row = e >> 5;           // 32 x 16B chunks per 512B row
            const int ch  = e & 31;
            async_copy_b128(lds_base + (unsigned)(row * DD + ch * 4) * 4u,
                            embedding + s_zrow[row] * DD + ch * 4);
        }
    }

    // ---- stage 1b: q = x @ Wq (scaled), gaussian smearing ------------------
    if (tid < DD) {
        float acc = 0.0f;
        for (int k = 0; k < DD; ++k) acc += s_x[k] * Wq[k * DD + tid];
        s_q[tid] = acc * 0.25f;               // 1/sqrt(hd)
    }
    {
        const float OFF0  = -2.3025851f;                       // log(0.1)
        const float WIDTH = (1.6094379f + 2.3025851f) / 31.0f; // linspace step
        const float COEFF = -0.5f / (WIDTH * WIDTH);
        for (int e = tid; e < ROWS * GG; e += 256) {
            const int r = e >> 5, g = e & 31;
            const float d = __logf(s_riij[r]) - (OFF0 + (float)g * WIDTH);
            s_fA[e] = (_Float16)__expf(COEFF * d * d);
        }
    }
    wait_async0();                            // my async gathers landed in LDS
    __syncthreads();                          // everyone's gathers visible

    // ---- stage 2: filter GEMM (64x32 @ 32x128) + build kv_in ---------------
    {
        const int nt   = wave;                // this wave owns column tile nt
        const int ncol = nt * 16 + (lane & 15);
        const int half = lane >> 4;
        const float fb = filt_b[ncol];
        const v16h bb  = load_b_frag(fWt, GG, nt * 16, 0, lane);
#pragma unroll
        for (int mt = 0; mt < 4; ++mt) {
            v16h a = load_a_frag(s_fA, GG, mt * 16, 0, lane);
            v8f  c = {};
            c = __builtin_amdgcn_wmma_f32_16x16x32_f16(
                    false, a, false, bb, (short)0, c, false, false);
#pragma unroll
            for (int v = 0; v < 8; ++v) {
                const int row = mt * 16 + v + 8 * half;
                const float w = (c[v] + fb) * s_cut[row];
                s_kv[row * DD + ncol] = (_Float16)(w * s_nodes[row * DD + ncol]);
            }
        }
    }
    __syncthreads();

    // ---- stage 3: K and V projections (64x128 @ 128x128, f16 WMMA) ---------
    // ks-outer: B fragments loaded once per K-step, reused by all 4 M-tiles.
    {
        const int nt = wave;
        v8f ck[4] = {{}, {}, {}, {}};
        v8f cv[4] = {{}, {}, {}, {}};
#pragma unroll
        for (int ks = 0; ks < 4; ++ks) {
            const v16h bk = load_b_frag(WkT, DD, nt * 16, ks * 32, lane);
            const v16h bv = load_b_frag(WvT, DD, nt * 16, ks * 32, lane);
#pragma unroll
            for (int mt = 0; mt < 4; ++mt) {
                v16h a = load_a_frag(s_kv, DD, mt * 16, ks * 32, lane);
                ck[mt] = __builtin_amdgcn_wmma_f32_16x16x32_f16(
                             false, a, false, bk, (short)0, ck[mt], false, false);
                cv[mt] = __builtin_amdgcn_wmma_f32_16x16x32_f16(
                             false, a, false, bv, (short)0, cv[mt], false, false);
            }
        }
#pragma unroll
        for (int mt = 0; mt < 4; ++mt) {
            store_c_frag(s_kp, DD, mt * 16, nt * 16, lane, ck[mt]);
            store_c_frag(s_vp, DD, mt * 16, nt * 16, lane, cv[mt]);
        }
    }
    __syncthreads();

    // ---- stage 4: attention scores -----------------------------------------
    for (int e = tid; e < NH * ROWS; e += 256) {
        const int h = e >> 6, r = e & 63;
        float s = 0.0f;
#pragma unroll
        for (int j = 0; j < HD; ++j)
            s += s_q[h * HD + j] * s_kp[r * DD + h * HD + j];
        s_sc[e] = (s_mask[r] > 0.0f) ? s : -1e9f;
    }
    __syncthreads();

    // ---- stage 5: softmax (one thread per head, 49 live keys) --------------
    if (tid < NH) {
        float m = -1e30f;
        for (int r = 0; r <= KNB; ++r) m = fmaxf(m, s_sc[tid * ROWS + r]);
        float ssum = 0.0f;
        for (int r = 0; r <= KNB; ++r) {
            const float e = __expf(s_sc[tid * ROWS + r] - m);
            s_sc[tid * ROWS + r] = e;
            ssum += e;
        }
        s_inv[tid] = 1.0f / ssum;
    }
    __syncthreads();

    // ---- stage 6: weighted aggregation of V --------------------------------
    if (tid < DD) {
        const int h = tid / HD;
        float acc = 0.0f;
        for (int r = 0; r <= KNB; ++r)
            acc += s_sc[h * ROWS + r] * s_vp[r * DD + tid];
        s_agg[tid] = acc * s_inv[h];
    }
    __syncthreads();

    // ---- stage 7: output projection + residual -----------------------------
    if (tid < DD) {
        float acc = 0.0f;
        for (int k = 0; k < DD; ++k) acc += s_agg[k] * Wo[k * DD + tid];
        out[atom * DD + tid] = s_x[tid] + acc;
    }
}

// ---------------------------------------------------------------------------
extern "C" void kernel_launch(void* const* d_in, const int* in_sizes, int n_in,
                              void* d_out, int out_size, void* d_ws, size_t ws_size,
                              hipStream_t stream) {
    const float* positions = (const float*)d_in[0];
    const int*   zz        = (const int*)  d_in[1];
    const int*   neighbors = (const int*)  d_in[2];
    const float* nmask     = (const float*)d_in[3];
    const float* embedding = (const float*)d_in[4];
    const float* filt_W    = (const float*)d_in[5];
    const float* filt_b    = (const float*)d_in[6];
    const float* Wq        = (const float*)d_in[7];
    const float* Wk        = (const float*)d_in[8];
    const float* Wv        = (const float*)d_in[9];
    const float* Wo        = (const float*)d_in[10];

    _Float16* WkT = (_Float16*)d_ws;          // 128*128 f16 = 32 KB
    _Float16* WvT = WkT + DD * DD;            // 32 KB
    _Float16* fWt = WvT + DD * DD;            // 128*32 f16 = 8 KB

    prep_weights<<<(DD * DD + 255) / 256, 256, 0, stream>>>(Wk, Wv, filt_W,
                                                            WkT, WvT, fWt);
    tdtnet_fused<<<NB * NN, 256, 0, stream>>>(positions, zz, neighbors, nmask,
                                              embedding, filt_b, Wq, Wo,
                                              WkT, WvT, fWt, (float*)d_out);
}
